// CNNBiLSTMCRF_61778809586271
// MI455X (gfx1250) — matrix operand order repaired and checked by reading, not compiled, not run on hardware
//
#include <hip/hip_runtime.h>
#include <hip/hip_bf16.h>

// ---------------------------------------------------------------------------
// BiLSTM-CRF for MI455X (gfx1250).
// Core GEMMs run on v_wmma_f32_16x16x32_bf16. Recurrence parallelized over
// (batch-tile x direction) persistent workgroups; h/c state lives in LDS.
// Double-buffered A panel -> single barrier per timestep.
// ---------------------------------------------------------------------------

#define BB 128
#define LL 256
#define HD 200
#define HDP 208            // gate region padded to 16-multiple
#define NGP (4 * HDP)      // 832 packed gate columns
#define KH 224             // h region in A/K space (200 padded to 32-mult)
#define KX0 160            // layer-0 x region (130 padded)
#define KX1 416            // layer-1 x region (2*208)
#define KT0 (KH + KX0)     // 384
#define KT1 (KH + KX1)     // 640
#define NT 18

typedef __attribute__((ext_vector_type(16))) __bf16 v16bf;
typedef __attribute__((ext_vector_type(8)))  float  v8f;

union Frag16 { uint4 q[2]; v16bf v; };

__device__ __forceinline__ unsigned short f2bf(float f) {
  union { float f; unsigned int u; } c; c.f = f;
  unsigned int u = c.u;
  unsigned int r = (u + 0x7FFFu + ((u >> 16) & 1u)) >> 16;
  return (unsigned short)r;
}
__device__ __forceinline__ float bf2f(unsigned short h) {
  union { unsigned int u; float f; } c; c.u = ((unsigned int)h) << 16;
  return c.f;
}
__device__ __forceinline__ float sigm(float x) {
  return 1.f / (1.f + __expf(-x));
}
__device__ __forceinline__ v8f splat8(float x) {
  v8f v;
#pragma unroll
  for (int r = 0; r < 8; ++r) v[r] = x;
  return v;
}

// ---------------------------------------------------------------------------
// 1) Embedding gather + concat + pad -> bf16 x[B][L][KX0]
// ---------------------------------------------------------------------------
__global__ __launch_bounds__(KX0) void embed_kernel(
    const int* __restrict__ words, const int* __restrict__ caps,
    const float* __restrict__ wemb, const float* __restrict__ cemb,
    unsigned short* __restrict__ x) {
  const int idx = blockIdx.x;          // b*L + t
  const int k = threadIdx.x;           // 0..159
  const int w = words[idx];
  const int c = caps[idx];
  float v = 0.f;
  if (k < 100)       v = wemb[(size_t)w * 100 + k];
  else if (k < 130)  v = cemb[(size_t)c * 30 + (k - 100)];
  x[(size_t)idx * KX0 + k] = f2bf(v);
}

// ---------------------------------------------------------------------------
// 2) Pack [W_hh | W_ih] -> bf16 [NGP][Ktot], gate regions padded 200->208,
//    K padded to 32-multiples. remap=1 maps layer-1 input columns to the
//    208-padded h0cat layout.
// ---------------------------------------------------------------------------
__global__ void pack_w_kernel(const float* __restrict__ Wih,
                              const float* __restrict__ Whh,
                              unsigned short* __restrict__ dst,
                              int Kin, int Ktot, int remap) {
  const int np = blockIdx.x;            // 0..NGP-1
  const int gate = np / HDP, r = np % HDP;
  const bool valid = (r < HD);
  const int n = gate * HD + r;
  for (int k = threadIdx.x; k < Ktot; k += blockDim.x) {
    float v = 0.f;
    if (valid) {
      if (k < KH) {
        if (k < HD) v = Whh[(size_t)n * HD + k];
      } else {
        const int kx = k - KH;
        int src = -1;
        if (remap) {
          if (kx < HD) src = kx;
          else if (kx >= HDP && kx < HDP + HD) src = kx - (HDP - HD);
        } else if (kx < Kin) {
          src = kx;
        }
        if (src >= 0) v = Wih[(size_t)n * Kin + src];
      }
    }
    dst[(size_t)np * Ktot + k] = f2bf(v);
  }
}

__global__ void pack_bias_kernel(const float* __restrict__ bih,
                                 const float* __restrict__ bhh,
                                 float* __restrict__ dst) {
  const int np = blockIdx.x * blockDim.x + threadIdx.x;
  if (np >= NGP) return;
  const int gate = np / HDP, r = np % HDP;
  dst[np] = (r < HD) ? (bih[gate * HD + r] + bhh[gate * HD + r]) : 0.f;
}

// ---------------------------------------------------------------------------
// 3) Persistent LSTM direction kernel.
//    grid = (8 batch tiles, 2 directions), block = 416 threads (13 waves).
//    Wave w owns hd-tile w (16 hd columns x 4 gates) for its 16 batch rows.
//    A = [h | x_t] double-buffered in LDS; B = packed weights from L2.
//    One barrier per timestep: GEMM reads buf[p], cell writes h into
//    buf[1-p], next x-stage writes buf[1-p]'s disjoint x-region.
// ---------------------------------------------------------------------------
__global__ __launch_bounds__(416) void lstm_kernel(
    const unsigned short* __restrict__ xsrc, int Kx, int Ktot,
    const unsigned short* __restrict__ wcombF,
    const unsigned short* __restrict__ wcombB,
    const float* __restrict__ biasF, const float* __restrict__ biasB,
    const int* __restrict__ seq_len, unsigned short* __restrict__ outbuf) {
  extern __shared__ char smem[];
  unsigned short* sA0 = (unsigned short*)smem;                     // [16][Ktot]
  unsigned short* sA1 = sA0 + (size_t)16 * Ktot;                   // [16][Ktot]
  float* sC = (float*)(smem + (size_t)32 * Ktot * 2);              // [16][HDP]
  int* sSeq = (int*)(sC + 16 * HDP);                               // [16]

  const int tid  = threadIdx.x;
  const int lane = tid & 31;
  const int wave = tid >> 5;            // 0..12
  const int b0   = (int)blockIdx.x * 16;
  const int dir  = blockIdx.y;
  const unsigned short* wcomb = dir ? wcombB : wcombF;
  const float* bias = dir ? biasB : biasF;

  for (int i = tid; i < 16 * Ktot; i += 416) { sA0[i] = 0; sA1[i] = 0; }
  for (int i = tid; i < 16 * HDP; i += 416) sC[i] = 0.f;
  if (tid < 16) sSeq[tid] = seq_len[b0 + tid];
  __syncthreads();

  const int hdt   = wave;               // hd tile
  const int ncol  = lane & 15;
  const int rowA  = lane & 15;
  const int hi    = lane >> 4;          // 0/1
  const int koffA = hi << 4;            // A frag: halves {0..15} / {16..31}
  const int koffB = hi << 4;            // B frag: halves {0..15} / {16..31}
  const int mrow0 = hi << 3;            // C layout: row = r + 8*hi
  const int chunksPerRow = Kx >> 3;
  const int nChunks = 16 * chunksPerRow;
  const int nKt = Ktot >> 5;
  const int hd = hdt * 16 + ncol;
  const float bi0 = bias[0 * HDP + hd];
  const float bi1 = bias[1 * HDP + hd];
  const float bi2 = bias[2 * HDP + hd];
  const float bi3 = bias[3 * HDP + hd];

  for (int step = 0; step < LL; ++step) {
    unsigned short* sA  = (step & 1) ? sA1 : sA0;   // read panel
    unsigned short* sAn = (step & 1) ? sA0 : sA1;   // write panel (next h)

    // Stage x_t into the read panel (per-row index handles _rev for bwd;
    // ridx is involutive so input and output positions coincide).
    for (int c = tid; c < nChunks; c += 416) {
      const int m = c / chunksPerRow, cc = c - m * chunksPerRow;
      const int s = sSeq[m];
      const int tin = dir ? ((step < s) ? (s - 1 - step) : step) : step;
      const uint4 v =
          *(const uint4*)(xsrc + ((size_t)((b0 + m) * LL + tin)) * Kx + cc * 8);
      *(uint4*)(sA + (size_t)m * Ktot + KH + cc * 8) = v;
    }
    __syncthreads();   // sole barrier: panel p fully assembled (h + x)

    v8f acc0 = splat8(bi0), acc1 = splat8(bi1);
    v8f acc2 = splat8(bi2), acc3 = splat8(bi3);
    for (int kt = 0; kt < nKt; ++kt) {
      const int kb = kt * 32;
      Frag16 a;
      a.q[0] = *(const uint4*)(sA + (size_t)rowA * Ktot + kb + koffA);
      a.q[1] = *(const uint4*)(sA + (size_t)rowA * Ktot + kb + koffA + 8);
      const unsigned short* wk = wcomb + (size_t)(kb + koffB);
      {
        Frag16 bg;
        const unsigned short* p = wk + (size_t)(0 * HDP + hd) * Ktot;
        bg.q[0] = *(const uint4*)p; bg.q[1] = *(const uint4*)(p + 8);
        acc0 = __builtin_amdgcn_wmma_f32_16x16x32_bf16(false, a.v, false, bg.v,
                                                       (short)0, acc0, false, false);
      }
      {
        Frag16 bg;
        const unsigned short* p = wk + (size_t)(1 * HDP + hd) * Ktot;
        bg.q[0] = *(const uint4*)p; bg.q[1] = *(const uint4*)(p + 8);
        acc1 = __builtin_amdgcn_wmma_f32_16x16x32_bf16(false, a.v, false, bg.v,
                                                       (short)0, acc1, false, false);
      }
      {
        Frag16 bg;
        const unsigned short* p = wk + (size_t)(2 * HDP + hd) * Ktot;
        bg.q[0] = *(const uint4*)p; bg.q[1] = *(const uint4*)(p + 8);
        acc2 = __builtin_amdgcn_wmma_f32_16x16x32_bf16(false, a.v, false, bg.v,
                                                       (short)0, acc2, false, false);
      }
      {
        Frag16 bg;
        const unsigned short* p = wk + (size_t)(3 * HDP + hd) * Ktot;
        bg.q[0] = *(const uint4*)p; bg.q[1] = *(const uint4*)(p + 8);
        acc3 = __builtin_amdgcn_wmma_f32_16x16x32_bf16(false, a.v, false, bg.v,
                                                       (short)0, acc3, false, false);
      }
    }

    // Element-wise LSTM cell update straight out of the WMMA accumulators.
    // Writes go to the *other* panel and to exclusively-owned sC columns,
    // so no barrier is needed before the next iteration's staging.
    unsigned short hb[8];
#pragma unroll
    for (int r = 0; r < 8; ++r) {
      const int m = mrow0 + r;
      const float cold = sC[m * HDP + hd];
      const float ig = sigm(acc0[r]);
      const float fg = sigm(acc1[r]);
      const float gg = tanhf(acc2[r]);
      const float og = sigm(acc3[r]);
      const float cn = fg * cold + ig * gg;
      const float h = og * tanhf(cn);
      sC[m * HDP + hd] = cn;
      const unsigned short v = f2bf(h);
      sAn[(size_t)m * Ktot + hd] = v;   // feeds next step's A panel
      hb[r] = v;
    }
#pragma unroll
    for (int r = 0; r < 8; ++r) {
      const int m = mrow0 + r;
      const int s = sSeq[m];
      const int tout = dir ? ((step < s) ? (s - 1 - step) : step) : step;
      const unsigned short v = (tout < s) ? hb[r] : (unsigned short)0;
      outbuf[((size_t)((b0 + m) * LL + tout)) * KX1 + dir * HDP + hd] = v;
    }
  }
}

// ---------------------------------------------------------------------------
// 4) logits = fw@Wf.T + bf + bw@Wb.T + bb   (tiny; VALU)
// ---------------------------------------------------------------------------
__global__ __launch_bounds__(256) void logits_kernel(
    const unsigned short* __restrict__ h1, const float* __restrict__ Wf,
    const float* __restrict__ bf_, const float* __restrict__ Wb,
    const float* __restrict__ bb_, float* __restrict__ logits) {
  __shared__ float sWf[NT * HD];
  __shared__ float sWb[NT * HD];
  __shared__ float sb[2 * NT];
  for (int i = threadIdx.x; i < NT * HD; i += 256) { sWf[i] = Wf[i]; sWb[i] = Wb[i]; }
  if (threadIdx.x < NT) { sb[threadIdx.x] = bf_[threadIdx.x]; sb[NT + threadIdx.x] = bb_[threadIdx.x]; }
  __syncthreads();
  const int idx = blockIdx.x * 256 + threadIdx.x;   // b*L + t
  const unsigned short* row = h1 + (size_t)idx * KX1;
  float acc[NT];
#pragma unroll
  for (int j = 0; j < NT; ++j) acc[j] = sb[j] + sb[NT + j];
  for (int k = 0; k < HD; ++k) {
    const float xf = bf2f(row[k]);
    const float xb = bf2f(row[HDP + k]);
#pragma unroll
    for (int j = 0; j < NT; ++j) acc[j] += xf * sWf[j * HD + k] + xb * sWb[j * HD + k];
  }
  for (int j = 0; j < NT; ++j) logits[(size_t)idx * NT + j] = acc[j];
}

// ---------------------------------------------------------------------------
// 5) CRF NLL: one wave per batch row; alpha in lanes 0..17; shfl broadcast.
// ---------------------------------------------------------------------------
__global__ __launch_bounds__(32) void crf_kernel(
    const float* __restrict__ logits, const int* __restrict__ target,
    const int* __restrict__ seq_len, const float* __restrict__ trans,
    float* __restrict__ out) {
  __shared__ float st[NT * NT];
  const int b = blockIdx.x, lane = threadIdx.x;
  for (int i = lane; i < NT * NT; i += 32) st[i] = trans[i];
  __syncthreads();
  const float NEG = -1e30f;
  const int s = seq_len[b];
  const int col = (lane < NT) ? lane : 0;
  float alpha = (lane < NT) ? logits[((size_t)b * LL) * NT + lane] : NEG;
  for (int t = 1; t < LL; ++t) {
    const float emit = (lane < NT) ? logits[((size_t)(b * LL + t)) * NT + lane] : 0.f;
    float v[NT];
    float m = NEG;
#pragma unroll
    for (int i = 0; i < NT; ++i) {
      const float ai = __shfl(alpha, i, 32);
      v[i] = ai + st[i * NT + col];
      m = fmaxf(m, v[i]);
    }
    float sum = 0.f;
#pragma unroll
    for (int i = 0; i < NT; ++i) sum += __expf(v[i] - m);
    const float nw = m + __logf(sum) + emit;
    if (lane < NT && t < s) alpha = nw;
  }
  // z = logsumexp over lanes 0..17
  float m = alpha;
  for (int off = 16; off > 0; off >>= 1) m = fmaxf(m, __shfl_xor(m, off, 32));
  float e = (lane < NT) ? __expf(alpha - m) : 0.f;
  for (int off = 16; off > 0; off >>= 1) e += __shfl_xor(e, off, 32);
  const float z = m + __logf(e);
  if (lane == 0) {
    float gold = 0.f;
    int prev = target[(size_t)b * LL];
    gold += logits[((size_t)b * LL) * NT + prev];   // t=0 always masked-in (s >= L/2)
    for (int t = 1; t < LL; ++t) {
      const int tg = target[(size_t)b * LL + t];
      if (t < s) {
        gold += logits[((size_t)(b * LL + t)) * NT + tg];
        gold += st[prev * NT + tg];
      }
      prev = tg;
    }
    out[b] = z - gold;
  }
}

// ---------------------------------------------------------------------------
// Host launcher
// ---------------------------------------------------------------------------
extern "C" void kernel_launch(void* const* d_in, const int* in_sizes, int n_in,
                              void* d_out, int out_size, void* d_ws, size_t ws_size,
                              hipStream_t stream) {
  (void)in_sizes; (void)n_in; (void)out_size; (void)ws_size;
  const int*   words   = (const int*)d_in[0];
  const int*   caps    = (const int*)d_in[1];
  const int*   seq_len = (const int*)d_in[2];
  const int*   target  = (const int*)d_in[3];
  const float* wemb    = (const float*)d_in[4];
  const float* cemb    = (const float*)d_in[5];
  const float* W_ih[4] = {(const float*)d_in[6],  (const float*)d_in[10],
                          (const float*)d_in[14], (const float*)d_in[18]};
  const float* W_hh[4] = {(const float*)d_in[7],  (const float*)d_in[11],
                          (const float*)d_in[15], (const float*)d_in[19]};
  const float* b_ih[4] = {(const float*)d_in[8],  (const float*)d_in[12],
                          (const float*)d_in[16], (const float*)d_in[20]};
  const float* b_hh[4] = {(const float*)d_in[9],  (const float*)d_in[13],
                          (const float*)d_in[17], (const float*)d_in[21]};
  const float* Wf  = (const float*)d_in[22];
  const float* bf_ = (const float*)d_in[23];
  const float* Wb  = (const float*)d_in[24];
  const float* bb_ = (const float*)d_in[25];
  const float* trans = (const float*)d_in[26];
  float* out = (float*)d_out;

  char* ws = (char*)d_ws;
  size_t off = 0;
  auto alloc = [&](size_t bytes) -> void* {
    void* p = ws + off;
    off = (off + bytes + 255) & ~(size_t)255;
    return p;
  };

  unsigned short* x    = (unsigned short*)alloc((size_t)BB * LL * KX0 * 2);
  unsigned short* w0f  = (unsigned short*)alloc((size_t)NGP * KT0 * 2);
  unsigned short* w0b  = (unsigned short*)alloc((size_t)NGP * KT0 * 2);
  unsigned short* w1f  = (unsigned short*)alloc((size_t)NGP * KT1 * 2);
  unsigned short* w1b  = (unsigned short*)alloc((size_t)NGP * KT1 * 2);
  float* bias0f = (float*)alloc((size_t)NGP * 4);
  float* bias0b = (float*)alloc((size_t)NGP * 4);
  float* bias1f = (float*)alloc((size_t)NGP * 4);
  float* bias1b = (float*)alloc((size_t)NGP * 4);
  unsigned short* h0 = (unsigned short*)alloc((size_t)BB * LL * KX1 * 2);
  unsigned short* h1 = (unsigned short*)alloc((size_t)BB * LL * KX1 * 2);
  float* logits = (float*)alloc((size_t)BB * LL * NT * 4);

  // 1) embeddings
  embed_kernel<<<BB * LL, KX0, 0, stream>>>(words, caps, wemb, cemb, x);

  // 2) weight / bias packing
  pack_w_kernel<<<NGP, 128, 0, stream>>>(W_ih[0], W_hh[0], w0f, 130, KT0, 0);
  pack_w_kernel<<<NGP, 128, 0, stream>>>(W_ih[1], W_hh[1], w0b, 130, KT0, 0);
  pack_w_kernel<<<NGP, 128, 0, stream>>>(W_ih[2], W_hh[2], w1f, 400, KT1, 1);
  pack_w_kernel<<<NGP, 128, 0, stream>>>(W_ih[3], W_hh[3], w1b, 400, KT1, 1);
  pack_bias_kernel<<<(NGP + 255) / 256, 256, 0, stream>>>(b_ih[0], b_hh[0], bias0f);
  pack_bias_kernel<<<(NGP + 255) / 256, 256, 0, stream>>>(b_ih[1], b_hh[1], bias0b);
  pack_bias_kernel<<<(NGP + 255) / 256, 256, 0, stream>>>(b_ih[2], b_hh[2], bias1f);
  pack_bias_kernel<<<(NGP + 255) / 256, 256, 0, stream>>>(b_ih[3], b_hh[3], bias1b);

  // 3) BiLSTM layers (persistent WMMA recurrence, double-buffered A panel)
  const size_t smem0 = (size_t)32 * KT0 * 2 + (size_t)16 * HDP * 4 + 64;  // ~37.9 KB
  const size_t smem1 = (size_t)32 * KT1 * 2 + (size_t)16 * HDP * 4 + 64;  // ~54.3 KB
  lstm_kernel<<<dim3(8, 2), 416, smem0, stream>>>(x, KX0, KT0, w0f, w0b,
                                                  bias0f, bias0b, seq_len, h0);
  lstm_kernel<<<dim3(8, 2), 416, smem1, stream>>>(h0, KX1, KT1, w1f, w1b,
                                                  bias1f, bias1b, seq_len, h1);

  // 4) emission logits
  logits_kernel<<<(BB * LL) / 256, 256, 0, stream>>>(h1, Wf, bf_, Wb, bb_, logits);

  // 5) CRF negative log-likelihood
  crf_kernel<<<BB, 32, 0, stream>>>(logits, target, seq_len, trans, out);
}
